// LlamaAttention_85272280695220
// MI455X (gfx1250) — matrix operand order repaired and checked by reading, not compiled, not run on hardware
//
#include <hip/hip_runtime.h>

#define H    2048
#define SEQ  2048
#define BATCH 2
#define NH   16
#define HD   128

typedef __attribute__((ext_vector_type(16))) __bf16 v16bf;
typedef __attribute__((ext_vector_type(8)))  __bf16 v8bf;
typedef __attribute__((ext_vector_type(8)))  float  v8f;
typedef __attribute__((ext_vector_type(4)))  float  v4f;
typedef __attribute__((ext_vector_type(4)))  unsigned int uint4v;
typedef __attribute__((ext_vector_type(8)))  int    int8v;
typedef __attribute__((ext_vector_type(4)))  int    int4v;

#if __has_builtin(__builtin_amdgcn_tensor_load_to_lds)
#define HAVE_TDM 1
#else
#define HAVE_TDM 0
#endif

template <short N>
static __device__ __forceinline__ void tdm_wait() {
#if __has_builtin(__builtin_amdgcn_s_wait_tensorcnt)
  __builtin_amdgcn_s_wait_tensorcnt(N);
#else
  asm volatile("s_wait_tensorcnt %0" ::"i"(N));
#endif
}

#if HAVE_TDM
// 2D tile load via Tensor Data Mover: tile (tile_d0 x tile_d1) elements of 2B
// out of a row-major tensor with row stride stride_el, into LDS at lds_byte.
static __device__ __forceinline__ void tdm_tile_load(const __bf16* gsrc,
                                                     unsigned lds_byte,
                                                     unsigned tile_d0, unsigned tile_d1,
                                                     unsigned tensor_d0, unsigned tensor_d1,
                                                     unsigned long long stride_el) {
  unsigned long long ga = (unsigned long long)(size_t)gsrc;
  uint4v g0;
  g0[0] = 1u;                                            // count=1 valid descriptor
  g0[1] = lds_byte;                                      // lds_addr
  g0[2] = (unsigned)(ga & 0xffffffffull);                // global_addr[31:0]
  g0[3] = (unsigned)((ga >> 32) & 0x1ffffffull) | (2u << 30);  // addr[56:32] | type=2
  int8v g1;
  g1[0] = (int)(1u << 16);                               // data_size=1 -> 2 bytes
  g1[1] = (int)((tensor_d0 & 0xffffu) << 16);            // tensor_dim0[15:0] @ bits63:48
  g1[2] = (int)(((tensor_d0 >> 16) & 0xffffu) |
                ((tensor_d1 & 0xffffu) << 16));          // dim0 hi | dim1 lo
  g1[3] = (int)(((tensor_d1 >> 16) & 0xffffu) |
                ((tile_d0 & 0xffffu) << 16));            // dim1 hi | tile_dim0
  g1[4] = (int)(tile_d1 & 0xffffu);                      // tile_dim1 (tile_dim2 = 0)
  g1[5] = (int)(stride_el & 0xffffffffull);              // tensor_dim0_stride[31:0]
  g1[6] = (int)((stride_el >> 32) & 0xffffull);          // stride[47:32]
  g1[7] = 0;
  int4v g2 = {};
  int4v g3 = {};
#if defined(__clang_major__) && __clang_major__ >= 23
  int8v g4 = {};
  __builtin_amdgcn_tensor_load_to_lds(g0, g1, g2, g3, g4, 0);
#else
  __builtin_amdgcn_tensor_load_to_lds(g0, g1, g2, g3, 0);
#endif
}
#endif

static __device__ __forceinline__ v16bf pack16(v8bf lo, v8bf hi) {
  v16bf r;
#pragma unroll
  for (int j = 0; j < 8; ++j) { r[j] = lo[j]; r[j + 8] = hi[j]; }
  return r;
}

static __device__ __forceinline__ v8f wmma_bf16(v16bf a, v16bf b, v8f c) {
  return __builtin_amdgcn_wmma_f32_16x16x32_bf16(false, a, false, b, (short)0, c,
                                                 false, false);
}

// ---------------------------------------------------------------- cast f32->bf16 (x8)
__global__ __launch_bounds__(256) void cast_f32_to_bf16(const float* __restrict__ in,
                                                        __bf16* __restrict__ out,
                                                        long n8) {
  long i = ((long)blockIdx.x * 256 + threadIdx.x) * 8;
  if (i < n8) {
    v4f a = *(const v4f*)(in + i);
    v4f b = *(const v4f*)(in + i + 4);
    v8bf r;
#pragma unroll
    for (int j = 0; j < 4; ++j) { r[j] = (__bf16)a[j]; r[j + 4] = (__bf16)b[j]; }
    *(v8bf*)(out + i) = r;
  }
}

// ---------------------------------------------------------------- GEMM  C = A @ B^T
// A:[M,K] bf16, Bw:[N,K] bf16 (nn.Linear weight). Block tile 128x128, 256 thr
// (8 waves), wave tile 32x64 (2x4 WMMA frags). A/B tiles double-buffered in LDS,
// filled by the Tensor Data Mover (TENSORcnt) when available.
#define BM 128
#define BN 128
#define BK 32

template <bool BF16OUT>
__global__ __launch_bounds__(256) void gemm_bf16_nt(const __bf16* __restrict__ A,
                                                    const __bf16* __restrict__ Bw,
                                                    void* __restrict__ Cout,
                                                    int M, int N, int K) {
  __shared__ __align__(16) __bf16 asmem[2][BM][BK];  // 2 x 8 KB
  __shared__ __align__(16) __bf16 bsmem[2][BN][BK];  // 2 x 8 KB

  const int warp = threadIdx.x >> 5;
  const int lane = threadIdx.x & 31;
  const int half = lane >> 4;
  const int l16  = lane & 15;
  const int m0blk = blockIdx.y * BM;
  const int n0blk = blockIdx.x * BN;
  const int waveM = (warp & 3) * 32;   // 4 waves along M
  const int waveN = (warp >> 2) * 64;  // 2 waves along N

  v8f acc[2][4];
#pragma unroll
  for (int mi = 0; mi < 2; ++mi)
#pragma unroll
    for (int ni = 0; ni < 4; ++ni)
#pragma unroll
      for (int v = 0; v < 8; ++v) acc[mi][ni][v] = 0.f;

  auto stage_issue = [&](int s, int buf) {
    const int k0 = s * BK;
#if HAVE_TDM
    if (warp == 0) {
      tdm_tile_load(A + (long)m0blk * K + k0,
                    (unsigned)(size_t)&asmem[buf][0][0],
                    BK, BM, (unsigned)K, (unsigned)M, (unsigned long long)K);
      tdm_tile_load(Bw + (long)n0blk * K + k0,
                    (unsigned)(size_t)&bsmem[buf][0][0],
                    BK, BN, (unsigned)K, (unsigned)N, (unsigned long long)K);
    }
#else
    {
      const int t = threadIdx.x;       // 256 threads copy 128x32 = 4096 el per tile
      const int row = t >> 1;
      const int off = (t & 1) * 16;
      *(v16bf*)&asmem[buf][row][off] =
          *(const v16bf*)(A + (long)(m0blk + row) * K + k0 + off);
      *(v16bf*)&bsmem[buf][row][off] =
          *(const v16bf*)(Bw + (long)(n0blk + row) * K + k0 + off);
      __builtin_prefetch(A + (long)(m0blk + row) * K + k0 + off + BK, 0, 1);
      __builtin_prefetch(Bw + (long)(n0blk + row) * K + k0 + off + BK, 0, 1);
    }
#endif
  };

  const int nstages = K / BK;
  stage_issue(0, 0);

  for (int s = 0; s < nstages; ++s) {
    const int cur = s & 1;
    if (s + 1 < nstages) stage_issue(s + 1, cur ^ 1);
#if HAVE_TDM
    if (warp == 0) {
      if (s + 1 < nstages) tdm_wait<2>();  // stage s complete (in-order TENSORcnt)
      else                 tdm_wait<0>();
    }
#endif
    __syncthreads();

    // consume current buffer: one K-step of 32
    v16bf af[2], bfr[4];
#pragma unroll
    for (int mi = 0; mi < 2; ++mi) {
      const __bf16* ar = &asmem[cur][waveM + mi * 16 + l16][0];
      af[mi] = pack16(*(const v8bf*)(ar + half * 16),
                      *(const v8bf*)(ar + half * 16 + 8));
    }
#pragma unroll
    for (int ni = 0; ni < 4; ++ni) {
      const __bf16* br = &bsmem[cur][waveN + ni * 16 + l16][0];
      bfr[ni] = *(const v16bf*)(br + half * 16);
    }
#pragma unroll
    for (int mi = 0; mi < 2; ++mi)
#pragma unroll
      for (int ni = 0; ni < 4; ++ni)
        acc[mi][ni] = wmma_bf16(af[mi], bfr[ni], acc[mi][ni]);

    __syncthreads();
  }

  // C/D layout: lane half h, vgpr v -> row h*8+v, col = l16
#pragma unroll
  for (int mi = 0; mi < 2; ++mi)
#pragma unroll
    for (int ni = 0; ni < 4; ++ni)
#pragma unroll
      for (int v = 0; v < 8; ++v) {
        long r = m0blk + waveM + mi * 16 + half * 8 + v;
        long c = n0blk + waveN + ni * 16 + l16;
        if (BF16OUT)
          ((__bf16*)Cout)[r * N + c] = (__bf16)acc[mi][ni][v];
        else
          ((float*)Cout)[r * N + c] = acc[mi][ni][v];
      }
}

// ---------------------------------------------------------------- RoPE (in place) + V transpose
__global__ __launch_bounds__(256) void rope_and_vt(__bf16* __restrict__ qb,
                                                   __bf16* __restrict__ kb,
                                                   const __bf16* __restrict__ vb,
                                                   __bf16* __restrict__ vT,
                                                   const int* __restrict__ pos_ids) {
  long t = (long)blockIdx.x * 256 + threadIdx.x;
  int i = (int)(t & 63);
  int h = (int)((t >> 6) & (NH - 1));
  int s = (int)((t >> 10) & (SEQ - 1));
  int b = (int)(t >> 21);
  long base = ((long)b * SEQ + s) * H + h * HD + i;

  float pos = (float)pos_ids[b * SEQ + s];
  float invf = __expf(-((float)i * (1.0f / 64.0f)) * 9.210340371976184f);
  float ang = pos * invf;
  float sn, cs;
  __sincosf(ang, &sn, &cs);

  float q0 = (float)qb[base], q1 = (float)qb[base + 64];
  qb[base]      = (__bf16)(q0 * cs - q1 * sn);
  qb[base + 64] = (__bf16)(q1 * cs + q0 * sn);
  float k0 = (float)kb[base], k1 = (float)kb[base + 64];
  kb[base]      = (__bf16)(k0 * cs - k1 * sn);
  kb[base + 64] = (__bf16)(k1 * cs + k0 * sn);

  long vt0 = ((long)(b * NH + h) * HD + i) * SEQ + s;
  vT[vt0]                  = vb[base];
  vT[vt0 + (long)64 * SEQ] = vb[base + 64];
}

// ---------------------------------------------------------------- flash attention
__global__ __launch_bounds__(128) void flash_attn(const __bf16* __restrict__ qb,
                                                  const __bf16* __restrict__ kb,
                                                  const __bf16* __restrict__ vT,
                                                  __bf16* __restrict__ ob) {
  __shared__ __align__(32) __bf16 plds[4][16][64];
  const int warp = threadIdx.x >> 5;
  const int lane = threadIdx.x & 31;
  const int half = lane >> 4;
  const int l16  = lane & 15;
  const int bh   = blockIdx.y;
  const int b    = bh >> 4;
  const int head = bh & 15;
  const int q0row = blockIdx.x * 64 + warp * 16;
  const float scale = 0.08838834764831845f;  // 1/sqrt(128)

  v16bf qf[4];
  const __bf16* qrow = qb + ((long)b * SEQ + q0row + l16) * H + head * HD;
#pragma unroll
  for (int kc = 0; kc < 4; ++kc)
    qf[kc] = pack16(*(const v8bf*)(qrow + kc * 32 + half * 16),
                    *(const v8bf*)(qrow + kc * 32 + half * 16 + 8));

  v8f o[8];
  float mrow[8], lrow[8];
#pragma unroll
  for (int d = 0; d < 8; ++d)
#pragma unroll
    for (int v = 0; v < 8; ++v) o[d][v] = 0.f;
#pragma unroll
  for (int v = 0; v < 8; ++v) { mrow[v] = -1e30f; lrow[v] = 0.f; }

  const __bf16* kbase = kb + ((long)b * SEQ) * H + head * HD;
  const __bf16* vbase = vT + (long)(b * NH + head) * HD * SEQ;

  for (int jb = 0; jb < SEQ / 64; ++jb) {
    v8f sfr[4];
#pragma unroll
    for (int nc = 0; nc < 4; ++nc)
#pragma unroll
      for (int v = 0; v < 8; ++v) sfr[nc][v] = 0.f;

#pragma unroll
    for (int nc = 0; nc < 4; ++nc) {
      const __bf16* krow = kbase + (long)(jb * 64 + nc * 16 + l16) * H;
      __builtin_prefetch(krow + (long)64 * H, 0, 0);  // next key block
#pragma unroll
      for (int kc = 0; kc < 4; ++kc) {
        v16bf bf = *(const v16bf*)(krow + kc * 32 + half * 16);
        sfr[nc] = wmma_bf16(qf[kc], bf, sfr[nc]);
      }
    }

    float corr[8];
#pragma unroll
    for (int v = 0; v < 8; ++v) {
      float mx = -1e30f;
#pragma unroll
      for (int nc = 0; nc < 4; ++nc) {
        sfr[nc][v] *= scale;
        mx = fmaxf(mx, sfr[nc][v]);
      }
      mx = fmaxf(mx, __shfl_xor(mx, 1, 32));
      mx = fmaxf(mx, __shfl_xor(mx, 2, 32));
      mx = fmaxf(mx, __shfl_xor(mx, 4, 32));
      mx = fmaxf(mx, __shfl_xor(mx, 8, 32));
      float mnew = fmaxf(mrow[v], mx);
      corr[v] = __expf(mrow[v] - mnew);
      mrow[v] = mnew;
      float rs = 0.f;
#pragma unroll
      for (int nc = 0; nc < 4; ++nc) {
        float p = __expf(sfr[nc][v] - mnew);
        sfr[nc][v] = p;
        rs += p;
      }
      rs += __shfl_xor(rs, 1, 32);
      rs += __shfl_xor(rs, 2, 32);
      rs += __shfl_xor(rs, 4, 32);
      rs += __shfl_xor(rs, 8, 32);
      lrow[v] = lrow[v] * corr[v] + rs;
    }
#pragma unroll
    for (int d = 0; d < 8; ++d)
#pragma unroll
      for (int v = 0; v < 8; ++v) o[d][v] *= corr[v];

#pragma unroll
    for (int nc = 0; nc < 4; ++nc)
#pragma unroll
      for (int v = 0; v < 8; ++v)
        plds[warp][half * 8 + v][nc * 16 + l16] = (__bf16)sfr[nc][v];

#pragma unroll
    for (int kc2 = 0; kc2 < 2; ++kc2) {
      const __bf16* prow = &plds[warp][l16][kc2 * 32];
      v16bf pa = pack16(*(const v8bf*)(prow + half * 16),
                        *(const v8bf*)(prow + half * 16 + 8));
#pragma unroll
      for (int d = 0; d < 8; ++d) {
        const __bf16* vrow = vbase + (long)(d * 16 + l16) * SEQ + jb * 64 + kc2 * 32;
        __builtin_prefetch(vrow + 64, 0, 0);
        v16bf bv = *(const v16bf*)(vrow + half * 16);
        o[d] = wmma_bf16(pa, bv, o[d]);
      }
    }
  }

#pragma unroll
  for (int v = 0; v < 8; ++v) {
    float invl = 1.0f / lrow[v];
    long srow = q0row + half * 8 + v;
    __bf16* orow = ob + ((long)b * SEQ + srow) * H + head * HD;
#pragma unroll
    for (int d = 0; d < 8; ++d)
      orow[d * 16 + l16] = (__bf16)(o[d][v] * invl);
  }
}

// ---------------------------------------------------------------- launch
extern "C" void kernel_launch(void* const* d_in, const int* in_sizes, int n_in,
                              void* d_out, int out_size, void* d_ws, size_t ws_size,
                              hipStream_t stream) {
  (void)in_sizes; (void)n_in; (void)out_size; (void)ws_size;

  const float* x   = (const float*)d_in[0];
  const int*   pos = (const int*)d_in[1];
  const float* Wq  = (const float*)d_in[2];
  const float* Wk  = (const float*)d_in[3];
  const float* Wv  = (const float*)d_in[4];
  const float* Wo  = (const float*)d_in[5];

  const long MROWS = (long)BATCH * SEQ;  // 4096
  const long NX = MROWS * H;
  const long NW = (long)H * H;

  char* ws = (char*)d_ws;
  __bf16* xb   = (__bf16*)ws; ws += NX * 2;
  __bf16* wqb  = (__bf16*)ws; ws += NW * 2;
  __bf16* wkb  = (__bf16*)ws; ws += NW * 2;
  __bf16* wvb  = (__bf16*)ws; ws += NW * 2;
  __bf16* wob  = (__bf16*)ws; ws += NW * 2;
  __bf16* qbuf = (__bf16*)ws; ws += NX * 2;
  __bf16* kbuf = (__bf16*)ws; ws += NX * 2;
  __bf16* vbuf = (__bf16*)ws; ws += NX * 2;
  __bf16* vTb  = (__bf16*)ws; ws += NX * 2;
  __bf16* abuf = (__bf16*)ws; ws += NX * 2;

  cast_f32_to_bf16<<<(int)((NX / 8 + 255) / 256), 256, 0, stream>>>(x,  xb,  NX);
  cast_f32_to_bf16<<<(int)((NW / 8 + 255) / 256), 256, 0, stream>>>(Wq, wqb, NW);
  cast_f32_to_bf16<<<(int)((NW / 8 + 255) / 256), 256, 0, stream>>>(Wk, wkb, NW);
  cast_f32_to_bf16<<<(int)((NW / 8 + 255) / 256), 256, 0, stream>>>(Wv, wvb, NW);
  cast_f32_to_bf16<<<(int)((NW / 8 + 255) / 256), 256, 0, stream>>>(Wo, wob, NW);

  dim3 gg(H / BN, (unsigned)(MROWS / BM));
  gemm_bf16_nt<true><<<gg, 256, 0, stream>>>(xb, wqb, qbuf, (int)MROWS, H, H);
  gemm_bf16_nt<true><<<gg, 256, 0, stream>>>(xb, wkb, kbuf, (int)MROWS, H, H);
  gemm_bf16_nt<true><<<gg, 256, 0, stream>>>(xb, wvb, vbuf, (int)MROWS, H, H);

  long nr = (long)BATCH * SEQ * NH * 64;
  rope_and_vt<<<(int)((nr + 255) / 256), 256, 0, stream>>>(qbuf, kbuf, vbuf, vTb, pos);

  dim3 ga(SEQ / 64, BATCH * NH);
  flash_attn<<<ga, 128, 0, stream>>>(qbuf, kbuf, vTb, abuf);

  gemm_bf16_nt<false><<<gg, 256, 0, stream>>>(abuf, wob, d_out, (int)MROWS, H, H);
}
